// TransformerDecoderLayer_21698174779945
// MI455X (gfx1250) — matrix operand order repaired and checked by reading, not compiled, not run on hardware
//
#include <hip/hip_runtime.h>
#include <stdint.h>

// ---------------------------------------------------------------------------
// Deformable-DETR decoder layer for MI455X (gfx1250, wave32).
// GEMMs: f16 operands staged to LDS with global_load_async_to_lds_b128
// (ASYNCcnt + s_wait_asynccnt, double-buffered), consumed by
// v_wmma_f32_16x16x32_f16. All operand buffers are padded so the async
// stage path has zero bounds logic (EXEC all-1s through WMMA).
// ---------------------------------------------------------------------------

typedef __attribute__((ext_vector_type(16))) _Float16 v16h;
typedef __attribute__((ext_vector_type(8)))  _Float16 v8h;
typedef __attribute__((ext_vector_type(8)))  float    v8f;
typedef __attribute__((ext_vector_type(4)))  int      v4i_;

#define BS_   32
#define LQ_   500
#define D_    256
#define NH_   8
#define HD_   32
#define FF_   1024
#define MROWS 16000
#define MPAD  16016     // +16 rows padding for attention tile overreach

#define BM 64
#define BN 128
#define BK 32

// ---- CDNA5 async global->LDS copy (16B per lane) ---------------------------
__device__ __forceinline__ void async_b128(void* lds, const void* g) {
#if defined(__gfx1250__)
#if __has_builtin(__builtin_amdgcn_global_load_async_to_lds_b128)
  __builtin_amdgcn_global_load_async_to_lds_b128(
      (__attribute__((address_space(1))) v4i_*)g,
      (__attribute__((address_space(3))) v4i_*)lds, 0, 0);
#else
  unsigned int l = (unsigned int)(__SIZE_TYPE__)(__attribute__((address_space(3))) void*)lds;
  unsigned long long a = (unsigned long long)(__SIZE_TYPE__)g;
  asm volatile("global_load_async_to_lds_b128 %0, %1, off" :: "v"(l), "v"(a) : "memory");
#endif
#endif
}
__device__ __forceinline__ void wait_async0() {
#if defined(__gfx1250__)
#if __has_builtin(__builtin_amdgcn_s_wait_asynccnt)
  __builtin_amdgcn_s_wait_asynccnt(0);
#else
  asm volatile("s_wait_asynccnt 0x0" ::: "memory");
#endif
#endif
}

// ---------------------------------------------------------------------------
// Batched WMMA GEMM, all-f16 operands:  C = epi( A[M,K] @ Bt[N,K]^T + bias )
//  - A row-major f16 (lda); Bt is the K-major "B^T" layout [n][k] (ldb).
//  - splitK/A2: A columns >= splitK come from A2 (concat without a copy).
//  - batch z = (bb, hh) with independent element strides.
//  - epi: 0 none, 1 relu, 2 sigmoid.  cf16: C stored as f16 instead of f32.
// 256 threads = 8 waves; block tile 64x128; wave = 16-row strip with four
// 16x16 f32 accumulators. Staging: 1 async b128 for A + 2 for B per thread
// per K-step, double buffered; one barrier per K-step.
// ---------------------------------------------------------------------------
__launch_bounds__(256)
__global__ void k_gemm16(const _Float16* __restrict__ A, const _Float16* __restrict__ A2, int splitK,
                         long long sAb, long long sAh, int lda,
                         const _Float16* __restrict__ Bt, long long sBb, long long sBh, int ldb,
                         const float* __restrict__ bias,
                         void* __restrict__ Cv, long long sCb, long long sCh, int ldc,
                         int M, int N, int K, int nH, int epi, int cf16)
{
  __shared__ __align__(16) _Float16 sA[2][BM * BK];   // [m][k]
  __shared__ __align__(16) _Float16 sB[2][BN * BK];   // [n][k]

  const int bz = blockIdx.z;
  const int bb = bz / nH, hh = bz - bb * nH;
  const _Float16* Ab  = A  + bb * sAb + hh * sAh;
  const _Float16* A2b = A2 + bb * sAb + hh * sAh;
  const _Float16* Bb  = Bt + bb * sBb + hh * sBh;

  const int row0 = blockIdx.y * BM;
  const int col0 = blockIdx.x * BN;
  const int tid  = threadIdx.x;
  const int wave = tid >> 5, lane = tid & 31;
  const int tr = wave & 3, tc0 = (wave >> 2) * 4;
  const int laneM = lane & 15;
  const int hiK8  = (lane >> 4) * 8;
  const int hiK16 = (lane >> 4) * 16;

  const int am  = tid >> 2;            // A tile row this thread stages
  const int ach = (tid & 3) * 8;       // half offset within row (16B chunk)
  const int bn  = tid >> 2;            // B tile rows bn and bn+64
  const int bch = (tid & 3) * 8;

  auto stage = [&](int buf, int k0) {
    const _Float16* ab; int kk;
    if (k0 < splitK) { ab = Ab; kk = k0; } else { ab = A2b; kk = k0 - splitK; }
    async_b128(&sA[buf][am * BK + ach],        ab + (long long)(row0 + am) * lda + kk + ach);
    async_b128(&sB[buf][bn * BK + bch],        Bb + (long long)(col0 + bn) * ldb + k0 + bch);
    async_b128(&sB[buf][(bn + 64) * BK + bch], Bb + (long long)(col0 + bn + 64) * ldb + k0 + bch);
  };

  v8f acc0 = {}, acc1 = {}, acc2 = {}, acc3 = {};
  const int nk = (K + BK - 1) / BK;
  stage(0, 0);
  for (int t = 0; t < nk; ++t) {
    wait_async0();        // our async copies for buffer t&1 are complete
    __syncthreads();      // ... and every wave's copies are visible
    if (t + 1 < nk) stage((t + 1) & 1, (t + 1) * BK);

    const _Float16* a = sA[t & 1];
    const _Float16* b = sB[t & 1];
    const int arow = (tr * 16 + laneM) * BK;
    v16h af = *(const v16h*)(a + arow + hiK16);

    v16h bf0 = *(const v16h*)(b + ((tc0 + 0) * 16 + laneM) * BK + hiK16);
    acc0 = __builtin_amdgcn_wmma_f32_16x16x32_f16(false, af, false, bf0, (short)0, acc0, false, false);
    v16h bf1 = *(const v16h*)(b + ((tc0 + 1) * 16 + laneM) * BK + hiK16);
    acc1 = __builtin_amdgcn_wmma_f32_16x16x32_f16(false, af, false, bf1, (short)0, acc1, false, false);
    v16h bf2 = *(const v16h*)(b + ((tc0 + 2) * 16 + laneM) * BK + hiK16);
    acc2 = __builtin_amdgcn_wmma_f32_16x16x32_f16(false, af, false, bf2, (short)0, acc2, false, false);
    v16h bf3 = *(const v16h*)(b + ((tc0 + 3) * 16 + laneM) * BK + hiK16);
    acc3 = __builtin_amdgcn_wmma_f32_16x16x32_f16(false, af, false, bf3, (short)0, acc3, false, false);
  }

  // epilogue: lane L -> (row = r + 8*(L>=16), col = L&15); masked stores
  const long long esz = cf16 ? 2 : 4;
  char* Cb = (char*)Cv + (bb * sCb + hh * sCh) * esz;
  const int rbase = row0 + tr * 16 + ((lane >> 4) * 8);
  v8f accs[4] = {acc0, acc1, acc2, acc3};
  #pragma unroll
  for (int t = 0; t < 4; ++t) {
    const int c = col0 + (tc0 + t) * 16 + laneM;
    if (c >= N) continue;
    const float bv = bias ? bias[c] : 0.f;
    #pragma unroll
    for (int r = 0; r < 8; ++r) {
      const int gr = rbase + r;
      if (gr >= M) continue;
      float v = accs[t][r] + bv;
      if (epi == 1)      v = v > 0.f ? v : 0.f;
      else if (epi == 2) v = 1.f / (1.f + __expf(-v));
      if (cf16) *(_Float16*)(Cb + ((long long)gr * ldc + c) * 2) = (_Float16)v;
      else      *(float*)   (Cb + ((long long)gr * ldc + c) * 4) = v;
    }
  }
}

// ---- softmax over attention scores: f32 in, f16 probs out (padding = 0) ----
__launch_bounds__(512)
__global__ void k_softmax(const float* __restrict__ S, _Float16* __restrict__ P,
                          int nvalid, int ld, float scale)
{
  __shared__ float red[512];
  const float* p = S + (long long)blockIdx.x * ld;
  _Float16* q = P + (long long)blockIdx.x * ld;
  const int t = threadIdx.x;
  float v = (t < nvalid) ? p[t] * scale : -3.4e38f;
  red[t] = v; __syncthreads();
  for (int s = 256; s > 0; s >>= 1) { if (t < s) red[t] = fmaxf(red[t], red[t + s]); __syncthreads(); }
  const float mx = red[0]; __syncthreads();
  float e = (t < nvalid) ? __expf(v - mx) : 0.f;
  red[t] = e; __syncthreads();
  for (int s = 256; s > 0; s >>= 1) { if (t < s) red[t] += red[t + s]; __syncthreads(); }
  q[t] = (_Float16)(e / red[0]);   // zero in padding -> K-tail safe for PV GEMM
}

// ---- fused LayerNorm over D=256 with optional f16 shadow output ------------
//  mode 0: v = x1+x2 ; mode 1: v = g1*x1+g2*x2 (gate ld 512) ; mode 2: clip
__launch_bounds__(256)
__global__ void k_ln(const float* __restrict__ x1, const float* __restrict__ x2,
                     const float* __restrict__ gate,
                     const float* __restrict__ g, const float* __restrict__ b,
                     float* __restrict__ out, _Float16* __restrict__ out16, int mode)
{
  __shared__ float red[256];
  const long long row = blockIdx.x;
  const int t = threadIdx.x;
  const float a = x1[row * D_ + t];
  const float c = x2[row * D_ + t];
  float v;
  if (mode == 0)      v = a + c;
  else if (mode == 1) v = gate[row * 512 + t] * a + gate[row * 512 + 256 + t] * c;
  else { v = a + c; v = fminf(fmaxf(v, -65504.f), 65504.f); }
  red[t] = v; __syncthreads();
  for (int s = 128; s > 0; s >>= 1) { if (t < s) red[t] += red[t + s]; __syncthreads(); }
  const float mean = red[0] * (1.f / D_); __syncthreads();
  const float d = v - mean;
  red[t] = d * d; __syncthreads();
  for (int s = 128; s > 0; s >>= 1) { if (t < s) red[t] += red[t + s]; __syncthreads(); }
  const float var = red[0] * (1.f / D_);
  const float o = d * rsqrtf(var + 1e-5f) * g[t] + b[t];
  out[row * D_ + t] = o;
  if (out16) out16[row * D_ + t] = (_Float16)o;
}

// ---- small elementwise helpers --------------------------------------------
__global__ void k_cvt(const float* __restrict__ a, _Float16* __restrict__ o, int n)
{
  const int i = blockIdx.x * blockDim.x + threadIdx.x;
  if (i < n) o[i] = (_Float16)a[i];
}
__global__ void k_add16(const float* __restrict__ a, const float* __restrict__ b,
                        _Float16* __restrict__ o, int n)
{
  const int i = blockIdx.x * blockDim.x + threadIdx.x;
  if (i < n) o[i] = (_Float16)(a[i] + b[i]);
}
// weight transpose: src f32 [K,N] -> dst f16 [n][k] (ldd = K)
__global__ void k_wtrans(const float* __restrict__ src, _Float16* __restrict__ dst,
                         int K, int N)
{
  const int i = blockIdx.x * blockDim.x + threadIdx.x;
  if (i >= N * K) return;
  const int n = i / K, k = i - n * K;
  dst[(long long)n * K + k] = (_Float16)src[(long long)k * N + n];
}
// V [16000,256] f16 -> Vt [bh][d(128 row alloc)][512] f16 (K-major for PV B)
__global__ void k_vtrans(const _Float16* __restrict__ v, _Float16* __restrict__ vt)
{
  const int i = blockIdx.x * blockDim.x + threadIdx.x;
  if (i >= 256 * HD_ * LQ_) return;
  const int q = i % LQ_;
  const int r = i / LQ_;
  const int d = r & 31, bh = r >> 5;
  const int b = bh >> 3, h = bh & 7;
  vt[((long long)bh * 128 + d) * 512 + q] = v[((long long)(b * LQ_ + q)) * D_ + h * HD_ + d];
}

// ---- deformable sampling: one wave per (b,q,h), 32 lanes = channels --------
__launch_bounds__(256)
__global__ void k_deform(const float* __restrict__ offs,  // [16000,192]
                         const float* __restrict__ awl,   // [16000, 96]
                         const float* __restrict__ ref,   // [16000,  4]
                         const float* __restrict__ v0, const float* __restrict__ v1,
                         const float* __restrict__ v2,
                         float* __restrict__ out, _Float16* __restrict__ out16)
{
  const int w    = (blockIdx.x * blockDim.x + threadIdx.x) >> 5;
  const int lane = threadIdx.x & 31;
  const int h   = w & 7;
  const int row = w >> 3;
  const int b   = row / LQ_;

  const float* awp = awl + (long long)row * 96 + h * 12;
  float awv[12];
  float mx = -3.4e38f;
  #pragma unroll
  for (int p = 0; p < 12; ++p) { awv[p] = awp[p]; mx = fmaxf(mx, awv[p]); }
  float s = 0.f;
  #pragma unroll
  for (int p = 0; p < 12; ++p) { awv[p] = __expf(awv[p] - mx); s += awv[p]; }
  const float inv = 1.f / s;

  const float* rp = ref + (long long)row * 4;
  const float cx = rp[0], cy = rp[1], rw = rp[2], rh = rp[3];
  const float* op = offs + (long long)row * 192 + h * 24;

  float acc = 0.f;
  #pragma unroll
  for (int p = 0; p < 12; ++p) {
    const int lvl = p >> 2;
    const float* vbase; int W, H;
    if (lvl == 0)      { vbase = v0; W = 80; H = 80; }
    else if (lvl == 1) { vbase = v1; W = 40; H = 40; }
    else               { vbase = v2; W = 20; H = 20; }
    const float lx = cx + op[p * 2 + 0] * 0.125f * rw;   // 1/NPTS * OFFSET_SCALE
    const float ly = cy + op[p * 2 + 1] * 0.125f * rh;
    const float px = lx * (float)W - 0.5f;               // grid_sample collapsed
    const float py = ly * (float)H - 0.5f;
    const float fx = floorf(px), fy = floorf(py);
    const int x0 = (int)fx, y0 = (int)fy;
    const float wx1 = px - fx, wx0 = 1.f - wx1;
    const float wy1 = py - fy, wy0 = 1.f - wy1;
    const float* base = vbase + ((long long)(b * NH_ + h) * HD_ + lane) * (long long)(W * H);
    const bool xin0 = (x0 >= 0) & (x0 < W);
    const bool xin1 = (x0 + 1 >= 0) & (x0 + 1 < W);
    const bool yin0 = (y0 >= 0) & (y0 < H);
    const bool yin1 = (y0 + 1 >= 0) & (y0 + 1 < H);
    const float t00 = (xin0 & yin0) ? base[y0 * W + x0]           : 0.f;
    const float t10 = (xin1 & yin0) ? base[y0 * W + x0 + 1]       : 0.f;
    const float t01 = (xin0 & yin1) ? base[(y0 + 1) * W + x0]     : 0.f;
    const float t11 = (xin1 & yin1) ? base[(y0 + 1) * W + x0 + 1] : 0.f;
    acc += (t00 * wx0 * wy0 + t10 * wx1 * wy0 + t01 * wx0 * wy1 + t11 * wx1 * wy1)
           * (awv[p] * inv);
  }
  const long long o = (long long)row * D_ + h * HD_ + lane;
  out[o] = acc;
  out16[o] = (_Float16)acc;
}

// ---------------------------------------------------------------------------
// Host side
// ---------------------------------------------------------------------------
static inline void gemm16(hipStream_t st,
                          const _Float16* A, const _Float16* A2, int splitK,
                          long long sAb, long long sAh, int lda,
                          const _Float16* Bt, long long sBb, long long sBh, int ldb,
                          const float* bias,
                          void* C, long long sCb, long long sCh, int ldc,
                          int M, int N, int K, int nH, int nB, int epi, int cf16)
{
  dim3 g((N + BN - 1) / BN, (M + BM - 1) / BM, nB * nH);
  k_gemm16<<<g, dim3(256), 0, st>>>(A, A2, splitK, sAb, sAh, lda,
                                    Bt, sBb, sBh, ldb, bias,
                                    C, sCb, sCh, ldc, M, N, K, nH, epi, cf16);
}

extern "C" void kernel_launch(void* const* d_in, const int* in_sizes, int n_in,
                              void* d_out, int out_size, void* d_ws, size_t ws_size,
                              hipStream_t stream)
{
  const float* target    = (const float*)d_in[0];
  const float* query_pos = (const float*)d_in[1];
  const float* ref_pts   = (const float*)d_in[2];
  const float* value0    = (const float*)d_in[3];
  const float* value1    = (const float*)d_in[4];
  const float* value2    = (const float*)d_in[5];
  const float* Wq  = (const float*)d_in[6];   const float* bq  = (const float*)d_in[7];
  const float* Wk  = (const float*)d_in[8];   const float* bk  = (const float*)d_in[9];
  const float* Wv  = (const float*)d_in[10];  const float* bv  = (const float*)d_in[11];
  const float* Wo  = (const float*)d_in[12];  const float* bo  = (const float*)d_in[13];
  const float* ln1g = (const float*)d_in[14]; const float* ln1b = (const float*)d_in[15];
  const float* Woff = (const float*)d_in[16]; const float* boff = (const float*)d_in[17];
  const float* Waw  = (const float*)d_in[18]; const float* baw  = (const float*)d_in[19];
  const float* Wg   = (const float*)d_in[20]; const float* bg   = (const float*)d_in[21];
  const float* lngg = (const float*)d_in[22]; const float* lngb = (const float*)d_in[23];
  const float* W1   = (const float*)d_in[24]; const float* b1   = (const float*)d_in[25];
  const float* W2   = (const float*)d_in[26]; const float* b2   = (const float*)d_in[27];
  const float* ln3g = (const float*)d_in[28]; const float* ln3b = (const float*)d_in[29];
  float* out = (float*)d_out;
  (void)in_sizes; (void)n_in; (void)out_size; (void)ws_size;

  // ---- workspace carve-up ----
  char* p = (char*)d_ws;
  auto af = [&](long long n) { float* r = (float*)p;    p += n * 4; return r; };
  auto ah = [&](long long n) { _Float16* r = (_Float16*)p; p += n * 2; return r; };

  float* RA = af((long long)MROWS * 256);   // t2 / t2f
  float* RB = af((long long)MROWS * 256);   // target1
  float* RC = af((long long)MROWS * 256);   // target2
  float* RD = af((long long)MROWS * 256);   // t2d
  float* RE = af((long long)MROWS * 192);   // offsets
  float* RF = af((long long)MROWS * 96);    // aw logits
  float* RG = af((long long)MROWS * 512);   // gate
  float* RS = af((long long)2 * NH_ * LQ_ * 512);       // scores chunk (f32)

  _Float16* hT  = ah((long long)MPAD * 256);  // target f16
  _Float16* hQK = ah((long long)MPAD * 256);  // target+pos
  _Float16* hQ  = ah((long long)MPAD * 256);
  _Float16* hK  = ah((long long)MPAD * 256);
  _Float16* hV  = ah((long long)MPAD * 256);
  _Float16* hVt = ah((long long)256 * 128 * 512);       // [bh][d][q] K-major
  _Float16* hP  = ah((long long)(2 * NH_ * LQ_ + 16) * 512);  // probs f16
  _Float16* hSA = ah((long long)MPAD * 256);
  _Float16* hQD = ah((long long)MPAD * 256);
  _Float16* hT1 = ah((long long)MPAD * 256);
  _Float16* hTD = ah((long long)MPAD * 256);
  _Float16* hT2 = ah((long long)MPAD * 256);
  _Float16* hH  = ah((long long)MPAD * 1024);
  _Float16* wq  = ah(256 * 256);
  _Float16* wk  = ah(256 * 256);
  _Float16* wv  = ah(256 * 256);
  _Float16* wo  = ah(256 * 256);
  _Float16* wof = ah(256 * 256);   // 192 valid rows, padded to 256
  _Float16* waw = ah(128 * 256);   // 96 valid rows, padded to 128
  _Float16* wg  = ah(512 * 512);
  _Float16* w1t = ah(1024 * 256);
  _Float16* w2t = ah(256 * 1024);

  const int nE = MROWS * 256;
  const float scl = 0.17677669529663687f;  // 1/sqrt(HD)

  // ---- one-time f16 conversions / weight transposes ----
  k_cvt<<<(nE + 255) / 256, 256, 0, stream>>>(target, hT, nE);
  k_add16<<<(nE + 255) / 256, 256, 0, stream>>>(target, query_pos, hQK, nE);
  k_wtrans<<<(256 * 256 + 255) / 256, 256, 0, stream>>>(Wq, wq, 256, 256);
  k_wtrans<<<(256 * 256 + 255) / 256, 256, 0, stream>>>(Wk, wk, 256, 256);
  k_wtrans<<<(256 * 256 + 255) / 256, 256, 0, stream>>>(Wv, wv, 256, 256);
  k_wtrans<<<(256 * 256 + 255) / 256, 256, 0, stream>>>(Wo, wo, 256, 256);
  k_wtrans<<<(192 * 256 + 255) / 256, 256, 0, stream>>>(Woff, wof, 256, 192);
  k_wtrans<<<(96 * 256 + 255) / 256, 256, 0, stream>>>(Waw, waw, 256, 96);
  k_wtrans<<<(512 * 512 + 255) / 256, 256, 0, stream>>>(Wg, wg, 512, 512);
  k_wtrans<<<(1024 * 256 + 255) / 256, 256, 0, stream>>>(W1, w1t, 256, 1024);
  k_wtrans<<<(256 * 1024 + 255) / 256, 256, 0, stream>>>(W2, w2t, 1024, 256);

  // ---- Q/K/V projections (f16 outputs) ----
  gemm16(stream, hQK, hQK, 256, 0,0, 256, wq, 0,0, 256, bq, hQ, 0,0, 256, MROWS, 256, 256, 1,1, 0, 1);
  gemm16(stream, hQK, hQK, 256, 0,0, 256, wk, 0,0, 256, bk, hK, 0,0, 256, MROWS, 256, 256, 1,1, 0, 1);
  gemm16(stream, hT,  hT,  256, 0,0, 256, wv, 0,0, 256, bv, hV, 0,0, 256, MROWS, 256, 256, 1,1, 0, 1);
  k_vtrans<<<(256 * HD_ * LQ_ + 255) / 256, 256, 0, stream>>>(hV, hVt);

  // ---- attention, chunked over batches (2 per chunk) ----
  const long long sQ = (long long)LQ_ * 256;
  for (int c0 = 0; c0 < BS_; c0 += 2) {
    // scores = Q @ K^T  (M=N=500, K=32, batched over 2*8)
    gemm16(stream, hQ + (long long)c0 * sQ, hQ, 32, sQ, HD_, 256,
           hK + (long long)c0 * sQ, sQ, HD_, 256, nullptr,
           RS, (long long)NH_ * LQ_ * 512, (long long)LQ_ * 512, 512,
           LQ_, LQ_, 32, NH_, 2, 0, 0);
    k_softmax<<<2 * NH_ * LQ_, 512, 0, stream>>>(RS, hP, LQ_, 512, scl);
    // sa = probs @ V  (M=500, N=32, K=500; probs K-tail zeroed by softmax)
    gemm16(stream, hP, hP, 512, (long long)NH_ * LQ_ * 512, (long long)LQ_ * 512, 512,
           hVt + (long long)c0 * NH_ * 128 * 512, (long long)NH_ * 128 * 512, (long long)128 * 512, 512,
           nullptr, hSA + (long long)c0 * sQ, sQ, HD_, 256,
           LQ_, HD_, LQ_, NH_, 2, 0, 1);
  }

  // ---- output projection + LN1 ----
  gemm16(stream, hSA, hSA, 256, 0,0, 256, wo, 0,0, 256, bo, RA, 0,0, 256, MROWS, 256, 256, 1,1, 0, 0);
  k_ln<<<MROWS, 256, 0, stream>>>(target, RA, nullptr, ln1g, ln1b, RB, hT1, 0);

  // ---- deformable branch ----
  k_add16<<<(nE + 255) / 256, 256, 0, stream>>>(RB, query_pos, hQD, nE);
  gemm16(stream, hQD, hQD, 256, 0,0, 256, wof, 0,0, 256, boff, RE, 0,0, 192, MROWS, 192, 256, 1,1, 0, 0);
  gemm16(stream, hQD, hQD, 256, 0,0, 256, waw, 0,0, 256, baw,  RF, 0,0,  96, MROWS,  96, 256, 1,1, 0, 0);
  k_deform<<<MROWS, 256, 0, stream>>>(RE, RF, ref_pts, value0, value1, value2, RD, hTD);

  // ---- gate (split-A concat, K=512, sigmoid) + gated LN ----
  gemm16(stream, hT1, hTD, 256, 0,0, 256, wg, 0,0, 512, bg, RG, 0,0, 512, MROWS, 512, 512, 1,1, 2, 0);
  k_ln<<<MROWS, 256, 0, stream>>>(RB, RD, RG, lngg, lngb, RC, hT2, 1);

  // ---- FFN + final LN ----
  gemm16(stream, hT2, hT2, 256, 0,0, 256, w1t, 0,0, 256, b1, hH, 0,0, 1024, MROWS, 1024, 256, 1,1, 1, 1);
  gemm16(stream, hH,  hH, 1024, 0,0, 1024, w2t, 0,0, 1024, b2, RA, 0,0, 256, MROWS, 256, 1024, 1,1, 0, 0);
  k_ln<<<MROWS, 256, 0, stream>>>(RC, RA, nullptr, ln3g, ln3b, out, nullptr, 2);
}
